// Net_69398081569013
// MI455X (gfx1250) — matrix operand order repaired and checked by reading, not compiled, not run on hardware
//
#include <hip/hip_runtime.h>

typedef __attribute__((ext_vector_type(16))) __bf16 v16bf;
typedef __attribute__((ext_vector_type(8)))  __bf16 v8bf;
typedef __attribute__((ext_vector_type(8)))  float  v8f;

// ---------------------------------------------------------------------------
// Degree / normalization: deg[i] = 1 (self loop) + #edges with target i
// ---------------------------------------------------------------------------
__global__ void deg_init_k(float* __restrict__ deg, int n) {
    int i = blockIdx.x * blockDim.x + threadIdx.x;
    if (i < n) deg[i] = 1.0f;
}

__global__ void deg_count_k(const int* __restrict__ col, float* __restrict__ deg, int E) {
    int e = blockIdx.x * blockDim.x + threadIdx.x;
    if (e < E) atomicAdd(&deg[col[e]], 1.0f);
}

__global__ void deg_rsqrt_k(float* __restrict__ deg, int n) {
    int i = blockIdx.x * blockDim.x + threadIdx.x;
    if (i < n) {
        float d = deg[i];
        deg[i] = (d > 0.0f) ? rsqrtf(d) : 0.0f;
    }
}

// ---------------------------------------------------------------------------
// Operand packing: f32 -> bf16 (row major), and f32 W[K,Nc] -> bf16 Wt[Nc,K]
// so WMMA B fragments are 16B runs along K.
// ---------------------------------------------------------------------------
__global__ void cvt_bf16_k(const float* __restrict__ in, __bf16* __restrict__ out, int n) {
    int i = blockIdx.x * blockDim.x + threadIdx.x;
    if (i < n) out[i] = (__bf16)in[i];
}

__global__ void cvt_transpose_bf16_k(const float* __restrict__ W, __bf16* __restrict__ Wt,
                                     int K, int Nc) {
    int idx = blockIdx.x * blockDim.x + threadIdx.x;
    if (idx < K * Nc) {
        int k = idx / Nc, n = idx - k * Nc;
        Wt[(size_t)n * K + k] = (__bf16)W[idx];
    }
}

// ---------------------------------------------------------------------------
// GEMM Out[M, NT*16] = Ab[M,K] @ Wt^T  (Wt stored [NT*16, K], bf16).
//
// Block = 4 waves. The whole weight panel Wt (NT*16 x K bf16) is staged in
// LDS once per block (coalesced b128 copies, one barrier), with row stride
// K+8 elements (68 dwords) so 16 consecutive rows map to disjoint bank
// groups -> conflict-free ds_load_b128 fragment reads.
// Each wave owns a 16-row stripe; A fragment (2x global b128) is reused
// across all NT column tiles; K,NT compile-time so all 4*NT WMMAs unroll.
//
// Fragment layouts (wave32, ISA 7.12.2):
//   A 16x32 bf16 : lane L -> row M=L&15; elems 0-7 = K[k0+8h .. +7],
//                  elems 8-15 = K[k0+16+8h .. +7]   (h = L>>4)
//   B 32x16 bf16 : lane L -> col N=L&15; elems 0-7 = K[k0+8h .. +7], elems 8-15 = K[k0+16+8h .. +7]
//   C/D f32      : VGPR r -> row r + 8h, col = L&15
// ---------------------------------------------------------------------------
template <int K, int NT>
__global__ void gemm_wmma_k(const __bf16* __restrict__ Ab,
                            const __bf16* __restrict__ Wt,
                            float* __restrict__ Out, int Mtiles) {
    constexpr int LDSK = K + 8;                       // padded row stride
    __shared__ __bf16 lw[NT * 16][LDSK];

    const int tid  = threadIdx.x;
    const int lane = tid & 31;
    const int half = lane >> 4;
    const int l16  = lane & 15;
    const int tm   = blockIdx.x * (blockDim.x >> 5) + (tid >> 5);

    // ---- cooperative stage of the weight panel into LDS (all waves) -----
    constexpr int CHUNKS = NT * 16 * (K / 8);         // 8-elem (16B) chunks
    for (int c = tid; c < CHUNKS; c += 128) {
        const int r  = c / (K / 8);
        const int c8 = (c - r * (K / 8)) * 8;
        *(v8bf*)&lw[r][c8] = *(const v8bf*)(Wt + (size_t)r * K + c8);
    }
    __syncthreads();

    if (tm < Mtiles) {
        const __bf16* arow = Ab + (size_t)(tm * 16 + l16) * K;

        v8f acc[NT];
#pragma unroll
        for (int t = 0; t < NT; ++t) acc[t] = (v8f){};

#pragma unroll
        for (int k0 = 0; k0 < K; k0 += 32) {
            const v8bf A0 = *(const v8bf*)(arow + k0 + 8 * half);
            const v8bf A1 = *(const v8bf*)(arow + k0 + 16 + 8 * half);
            v16bf a;
#pragma unroll
            for (int i = 0; i < 8; ++i) { a[i] = A0[i]; a[i + 8] = A1[i]; }

#pragma unroll
            for (int t = 0; t < NT; ++t) {
                const v8bf B0 = *(const v8bf*)&lw[t * 16 + l16][k0 + 8 * half];
                const v8bf B1 = *(const v8bf*)&lw[t * 16 + l16][k0 + 16 + 8 * half];
                v16bf b;
#pragma unroll
                for (int i = 0; i < 8; ++i) { b[i] = B0[i]; b[i + 8] = B1[i]; }
                acc[t] = __builtin_amdgcn_wmma_f32_16x16x32_bf16(
                             false, a, false, b, (short)0, acc[t], false, false);
            }
        }

        const int Nc = NT * 16;
#pragma unroll
        for (int t = 0; t < NT; ++t) {
#pragma unroll
            for (int r = 0; r < 8; ++r) {
                const int row = tm * 16 + half * 8 + r;
                Out[(size_t)row * Nc + t * 16 + l16] = acc[t][r];
            }
        }
    }
}

// ---------------------------------------------------------------------------
// Aggregation: out[i] = sum_{(j->i)} h[j]*dinv[j]*dinv[i] + h[i]*dinv[i]^2
// ---------------------------------------------------------------------------
__global__ void agg_self_k(const float* __restrict__ h, const float* __restrict__ dinv,
                           float* __restrict__ out, int n, int F) {
    int idx = blockIdx.x * blockDim.x + threadIdx.x;
    if (idx < n * F) {
        const int i = idx / F;
        const float di = dinv[i];
        out[idx] = h[idx] * di * di;
    }
}

__global__ void agg_edge_k(const float* __restrict__ h, const float* __restrict__ dinv,
                           const int* __restrict__ row, const int* __restrict__ col,
                           float* __restrict__ out, int E, int F) {
    int idx = blockIdx.x * blockDim.x + threadIdx.x;
    if (idx >= E * F) return;
    const int e = idx / F;
    const int f = idx - e * F;                // feat fastest -> coalesced gather+atomic
    const int j = row[e];
    const int i = col[e];
    const float nrm = dinv[i] * dinv[j];
    atomicAdd(&out[(size_t)i * F + f], h[(size_t)j * F + f] * nrm);
}

__global__ void bias_act_k(float* __restrict__ h, const float* __restrict__ b,
                           int n, int F, int do_relu) {
    int idx = blockIdx.x * blockDim.x + threadIdx.x;
    if (idx < n * F) {
        float v = h[idx] + b[idx % F];
        h[idx] = do_relu ? fmaxf(v, 0.0f) : v;
    }
}

// ---------------------------------------------------------------------------
// Edge scoring: one wave32 per edge, 64-dim dot product, shfl_xor reduction.
// ---------------------------------------------------------------------------
__global__ void edge_score_k(const float* __restrict__ h2,
                             const int* __restrict__ pos, const int* __restrict__ neg,
                             float* __restrict__ out, int Ep, int En, int F) {
    const int w    = blockIdx.x * (blockDim.x >> 5) + (threadIdx.x >> 5);
    const int lane = threadIdx.x & 31;
    if (w >= Ep + En) return;

    int s, d;
    if (w < Ep) { s = pos[w];       d = pos[Ep + w]; }
    else        { int e = w - Ep; s = neg[e]; d = neg[En + e]; }

    float acc = 0.0f;
    for (int f = lane; f < F; f += 32)
        acc += h2[(size_t)s * F + f] * h2[(size_t)d * F + f];

#pragma unroll
    for (int m = 16; m > 0; m >>= 1)
        acc += __shfl_xor(acc, m, 32);

    if (lane == 0) out[w] = acc;
}

// ---------------------------------------------------------------------------
extern "C" void kernel_launch(void* const* d_in, const int* in_sizes, int n_in,
                              void* d_out, int out_size, void* d_ws, size_t ws_size,
                              hipStream_t stream) {
    const float* x     = (const float*)d_in[0];
    const int*   train = (const int*)  d_in[1];
    const int*   pos   = (const int*)  d_in[2];
    const int*   neg   = (const int*)  d_in[3];
    const float* W1    = (const float*)d_in[4];
    const float* b1    = (const float*)d_in[5];
    const float* W2    = (const float*)d_in[6];
    const float* b2    = (const float*)d_in[7];
    float* out = (float*)d_out;

    const int N = 50000, Fin = 128, H1 = 128, H2 = 64;
    const int Etr = in_sizes[1] / 2;          // 600000
    const int Ep  = in_sizes[2] / 2;          // 200000
    const int En  = in_sizes[3] / 2;          // 200000
    const int* row = train;                   // sources  (edge_index[0])
    const int* col = train + Etr;             // targets  (edge_index[1])

    // ---- workspace carve (byte offsets, 256B-aligned) -------------------
    char* w = (char*)d_ws;
    float*  dinv = (float*)w;                              //   200,704 B
    __bf16* Xb   = (__bf16*)(w + 200704);                  // 12,800,000 B (N*128 bf16)
    __bf16* W1t  = (__bf16*)(w + 13000704);                //    32,768 B (128x128)
    __bf16* W2t  = (__bf16*)(w + 13033472);                //    16,384 B (64x128)
    float*  bufA = (float*)(w + 13049856);                 // 25,600,000 B (N*128 f32)
    float*  bufB = (float*)(w + 38649856);                 // 25,600,000 B (N*128 f32)

    // ---- normalization coefficients -------------------------------------
    deg_init_k <<<(N + 255) / 256, 256, 0, stream>>>(dinv, N);
    deg_count_k<<<(Etr + 255) / 256, 256, 0, stream>>>(col, dinv, Etr);
    deg_rsqrt_k<<<(N + 255) / 256, 256, 0, stream>>>(dinv, N);

    // ---- pack operands to bf16 ------------------------------------------
    cvt_bf16_k<<<(N * Fin + 255) / 256, 256, 0, stream>>>(x, Xb, N * Fin);
    cvt_transpose_bf16_k<<<(Fin * H1 + 255) / 256, 256, 0, stream>>>(W1, W1t, Fin, H1);
    cvt_transpose_bf16_k<<<(H1 * H2 + 255) / 256, 256, 0, stream>>>(W2, W2t, H1, H2);

    const int Mtiles = N / 16;                // 3125
    const int gemmBlocks = (Mtiles + 3) / 4;  // 4 waves (128 thr) per block

    // ---- layer 1: h1 = relu(aggregate(x@W1) + b1) -----------------------
    gemm_wmma_k<128, 8><<<gemmBlocks, 128, 0, stream>>>(Xb, W1t, bufA, Mtiles);
    agg_self_k<<<(N * H1 + 255) / 256, 256, 0, stream>>>(bufA, dinv, bufB, N, H1);
    agg_edge_k<<<(Etr * H1 + 255) / 256, 256, 0, stream>>>(bufA, dinv, row, col, bufB, Etr, H1);
    bias_act_k<<<(N * H1 + 255) / 256, 256, 0, stream>>>(bufB, b1, N, H1, 1);

    // ---- layer 2: h2 = aggregate(h1@W2) + b2 ----------------------------
    cvt_bf16_k<<<(N * H1 + 255) / 256, 256, 0, stream>>>(bufB, Xb, N * H1);
    gemm_wmma_k<128, 4><<<gemmBlocks, 128, 0, stream>>>(Xb, W2t, bufA, Mtiles);
    agg_self_k<<<(N * H2 + 255) / 256, 256, 0, stream>>>(bufA, dinv, bufB, N, H2);
    agg_edge_k<<<(Etr * H2 + 255) / 256, 256, 0, stream>>>(bufA, dinv, row, col, bufB, Etr, H2);
    bias_act_k<<<(N * H2 + 255) / 256, 256, 0, stream>>>(bufB, b2, N, H2, 0);

    // ---- edge scores ----------------------------------------------------
    const int Etot = Ep + En;
    edge_score_k<<<(Etot * 32 + 255) / 256, 256, 0, stream>>>(bufB, pos, neg, out, Ep, En, H2);
}
